// MultiHeadSelfAttention_12068858102016
// MI455X (gfx1250) — matrix operand order repaired and checked by reading, not compiled, not run on hardware
//
#include <hip/hip_runtime.h>
#include <hip/hip_bf16.h>

typedef unsigned short u16;
typedef unsigned int u32;
typedef __attribute__((ext_vector_type(16))) __bf16 v16bf;
typedef __attribute__((ext_vector_type(8)))  float  v8f;
typedef int gv4i __attribute__((vector_size(16)));   // matches async-LDS builtin param type

#define WMMA_BF16(a,b,c) \
  __builtin_amdgcn_wmma_f32_16x16x32_bf16(false,(a),false,(b),(short)0,(c),false,false)

#define AS1 __attribute__((address_space(1)))
#define AS3 __attribute__((address_space(3)))

#if defined(__has_builtin)
#if __has_builtin(__builtin_amdgcn_global_load_async_to_lds_b128) && \
    __has_builtin(__builtin_amdgcn_s_wait_asynccnt)
#define HAVE_ASYNC_LDS 1
#endif
#if __has_builtin(__builtin_amdgcn_permlane16)
#define HAVE_PERMLANE16 1
#endif
#endif

static constexpr int Bsz = 4, Tn = 2048, Cn = 1024, Hn = 16, Dn = 64, N3 = 3072;

__device__ __forceinline__ u16 f2bf(float f) {
  u32 u = __builtin_bit_cast(u32, f);
  u32 r = u + 0x7FFFu + ((u >> 16) & 1u);   // round-to-nearest-even
  return (u16)(r >> 16);
}

// xor-butterfly within each 16-lane half, on the VALU (v_permlane16_b32) when available
__device__ __forceinline__ float xsh(float x, int m, u32 s0, u32 s1) {
#ifdef HAVE_PERMLANE16
  int xi = __builtin_bit_cast(int, x);
  int r  = __builtin_amdgcn_permlane16(xi, xi, (int)s0, (int)s1, false, false);
  return __builtin_bit_cast(float, r);
#else
  return __shfl_xor(x, m, 32);
#endif
}
// sel tables: lane i reads lane i^m (nibble i of {s1,s0} = source lane within 16-group)
#define XMAX16(mx) do {                                       \
    mx = fmaxf(mx, xsh(mx, 1, 0x67452301u, 0xEFCDAB89u));     \
    mx = fmaxf(mx, xsh(mx, 2, 0x54761032u, 0xDCFE98BAu));     \
    mx = fmaxf(mx, xsh(mx, 4, 0x32107654u, 0xBA98FEDCu));     \
    mx = fmaxf(mx, xsh(mx, 8, 0xFEDCBA98u, 0x76543210u));     \
  } while (0)

union FragU { v16bf v; uint4 q[2]; u16 s[16]; };
union LdU   { uint4 q; u16 s[8]; };

// ---------------- fp32 -> bf16 conversion ----------------
__global__ void cvt_f32_bf16(const float* __restrict__ in, u16* __restrict__ out, int n) {
  int i = blockIdx.x * blockDim.x + threadIdx.x;
  if (i < n) out[i] = f2bf(in[i]);
}

// ---------------- bf16 GEMM: C[M,N] = A[M,K] * B[K,N] ----------------
// Block tile 128x128, 8 waves (4 in M x 2 in N), wave tile 32x64.
// MODE 0: QKV epilogue (scatter to per-head Q (scaled), K, V^T); MODE 1: f32 out.
template<int MODE>
__global__ __launch_bounds__(256)
void gemm_bf16(const u16* __restrict__ A, const u16* __restrict__ Bm,
               int Kdim, int Nn,
               u16* __restrict__ Qb, u16* __restrict__ Kb, u16* __restrict__ Vt,
               float* __restrict__ Out)
{
  __shared__ u16 As[128 * 32];    // row-major [128][32]
  __shared__ u16 Bst[128 * 32];   // transposed: [col 0..127][k 0..31]

  const int tid   = threadIdx.x;
  const int lane  = tid & 31;
  const int wave  = tid >> 5;
  const int wm    = wave & 3;
  const int wn    = wave >> 2;
  const int rhalf = lane >> 4;
  const int chalf = lane & 15;
  const int base1 = rhalf * 8;
  const int mblock = blockIdx.x;
  const int nblock = blockIdx.y;

  v8f acc[2][4];
#pragma unroll
  for (int mt = 0; mt < 2; mt++)
#pragma unroll
    for (int nt = 0; nt < 4; nt++)
#pragma unroll
      for (int r = 0; r < 8; r++) acc[mt][nt][r] = 0.f;

  for (int k0 = 0; k0 < Kdim; k0 += 32) {
    // ---- Stage A tile (128x32 bf16), async DMA into LDS ----
    for (int i = tid; i < 512; i += 256) {
      int row = i >> 2, ch = i & 3;
      const u16* g = &A[(size_t)(mblock * 128 + row) * Kdim + k0 + ch * 8];
#ifdef HAVE_ASYNC_LDS
      __builtin_amdgcn_global_load_async_to_lds_b128(
          (AS1 gv4i*)(unsigned long long)(size_t)g,
          (AS3 gv4i*)(u32)(size_t)&As[row * 32 + ch * 8],
          0, 0);
#else
      *(uint4*)&As[row * 32 + ch * 8] = *(const uint4*)g;
#endif
    }
    // ---- Stage B tile (32x128) transposed into Bst[col][k] ----
    for (int i = tid; i < 512; i += 256) {
      int kr = i >> 4, ch = i & 15;
      LdU u;
      u.q = *(const uint4*)&Bm[(size_t)(k0 + kr) * Nn + nblock * 128 + ch * 8];
#pragma unroll
      for (int e = 0; e < 8; e++) Bst[(ch * 8 + e) * 32 + kr] = u.s[e];
    }
    // Prefetch next K-step tiles (global_prefetch_b8; speculative, drop-on-fault)
    if (k0 + 32 < Kdim) {
      __builtin_prefetch(&A[(size_t)(mblock * 128 + (tid >> 1)) * Kdim + k0 + 32 + (tid & 1) * 16], 0, 1);
      __builtin_prefetch(&Bm[(size_t)(k0 + 32 + (tid >> 4)) * Nn + nblock * 128 + (tid & 15) * 8], 0, 1);
    }
#ifdef HAVE_ASYNC_LDS
    __builtin_amdgcn_s_wait_asynccnt(0);
#endif
    __syncthreads();

    FragU af[2], bf[4];
#pragma unroll
    for (int mt = 0; mt < 2; mt++) {
      int row = wm * 32 + mt * 16 + chalf;
      af[mt].q[0] = *(const uint4*)&As[row * 32 + rhalf * 16];
      af[mt].q[1] = *(const uint4*)&As[row * 32 + rhalf * 16 + 8];
    }
#pragma unroll
    for (int nt = 0; nt < 4; nt++) {
      int col = wn * 64 + nt * 16 + chalf;
      bf[nt].q[0] = *(const uint4*)&Bst[col * 32 + rhalf * 16];
      bf[nt].q[1] = *(const uint4*)&Bst[col * 32 + rhalf * 16 + 8];
    }
#pragma unroll
    for (int mt = 0; mt < 2; mt++)
#pragma unroll
      for (int nt = 0; nt < 4; nt++)
        acc[mt][nt] = WMMA_BF16(af[mt].v, bf[nt].v, acc[mt][nt]);
    __syncthreads();
  }

  // Epilogue. C/D layout: VGPR r, lane -> (M = r + 8*(lane>>4), N = lane&15)
#pragma unroll
  for (int mt = 0; mt < 2; mt++)
#pragma unroll
    for (int nt = 0; nt < 4; nt++)
#pragma unroll
      for (int r = 0; r < 8; r++) {
        int gm = mblock * 128 + wm * 32 + mt * 16 + r + 8 * rhalf;
        int gn = nblock * 128 + wn * 64 + nt * 16 + chalf;
        float val = acc[mt][nt][r];
        if (MODE == 0) {
          int b = gm / Tn, t = gm % Tn;
          int which = gn / Cn, c = gn % Cn;
          int h = c / Dn, dd = c % Dn;
          size_t bh = (size_t)(b * Hn + h);
          if (which == 0)      Qb[(bh * Tn + t) * Dn + dd] = f2bf(val * 0.125f); // 1/sqrt(64)
          else if (which == 1) Kb[(bh * Tn + t) * Dn + dd] = f2bf(val);
          else                 Vt[(bh * Dn + dd) * Tn + t] = f2bf(val);          // V transposed
        } else {
          Out[(size_t)gm * Nn + gn] = val;
        }
      }
}

// ---------------- Flash-attention tile: 16 query rows x 32 keys ----------------
// o[0..3] = output accumulators; o[4] = row-sum accumulator (ones-column trick:
// the exp row-sum is computed by the matrix pipe via an all-ones B fragment).
template<bool CAUSAL>
__device__ __forceinline__ void attn_tile(
    const u16* __restrict__ Kb, const u16* __restrict__ Vt,
    const unsigned char* __restrict__ tok,
    const FragU* qa, const FragU& ones, v8f* o, float* mrow, u16* P,
    int bh, int b, int qrow, int jbase, int lane)
{
  const int rhalf = lane >> 4;
  const int chalf = lane & 15;
  const int base1 = rhalf * 8;

  v8f s0, s1;
#pragma unroll
  for (int r = 0; r < 8; r++) { s0[r] = 0.f; s1[r] = 0.f; }

  // S = Q * K^T  (B-fragment of K^T is a contiguous 32B read of a key row)
#pragma unroll
  for (int kc = 0; kc < 2; kc++) {
    FragU kb0, kb1;
    const u16* kp0 = Kb + ((size_t)bh * Tn + jbase + chalf) * Dn + kc * 32 + rhalf * 16;
    const u16* kp1 = kp0 + 16 * Dn;
    kb0.q[0] = *(const uint4*)kp0; kb0.q[1] = *(const uint4*)(kp0 + 8);
    kb1.q[0] = *(const uint4*)kp1; kb1.q[1] = *(const uint4*)(kp1 + 8);
    s0 = WMMA_BF16(qa[kc].v, kb0.v, s0);
    s1 = WMMA_BF16(qa[kc].v, kb1.v, s1);
  }

  // Token mask as additive bias (causal compare only in diagonal tiles)
  const int c0 = jbase + chalf, c1 = c0 + 16;
  const float bias0 = (tok[b * Tn + c0] != 0) ? 0.f : -1.0e9f;
  const float bias1 = (tok[b * Tn + c1] != 0) ? 0.f : -1.0e9f;

#pragma unroll
  for (int r = 0; r < 8; r++) {
    float v0 = s0[r] + bias0;
    float v1 = s1[r] + bias1;
    if (CAUSAL) {
      int row = qrow + r + 8 * rhalf;
      v0 = (c0 <= row) ? v0 : -1.0e9f;
      v1 = (c1 <= row) ? v1 : -1.0e9f;
    }
    float mx = fmaxf(v0, v1);
    XMAX16(mx);
    float mnew = fmaxf(mrow[r], mx);
    float corr = __expf(mrow[r] - mnew);
    float p0 = __expf(v0 - mnew);
    float p1 = __expf(v1 - mnew);
    mrow[r] = mnew;
#pragma unroll
    for (int a = 0; a < 5; a++) o[a][r] *= corr;   // includes row-sum accumulator
    int prow = r + 8 * rhalf;
    P[prow * 32 + chalf]      = f2bf(p0);
    P[prow * 32 + 16 + chalf] = f2bf(p1);
  }

  // Re-fragment P (C-layout -> A-layout) via per-wave LDS
  FragU pa;
  {
    const u16* pp = &P[chalf * 32 + rhalf * 16];
    pa.q[0] = *(const uint4*)pp;
    pa.q[1] = *(const uint4*)(pp + 8);
  }

  // O += P * V  (V transposed [d][T] -> contiguous B-fragment reads)
#pragma unroll
  for (int nt = 0; nt < 4; nt++) {
    FragU vb;
    const u16* vp = Vt + ((size_t)bh * Dn + nt * 16 + chalf) * Tn + jbase + rhalf * 16;
    vb.q[0] = *(const uint4*)vp;
    vb.q[1] = *(const uint4*)(vp + 8);
    o[nt] = WMMA_BF16(pa.v, vb.v, o[nt]);
  }
  // Row sums of P on the matrix pipe (replaces 32 cross-lane adds per tile)
  o[4] = WMMA_BF16(pa.v, ones.v, o[4]);
}

// ---------------- Flash attention: one (b,h) x 128 query rows per block ----------------
__global__ __launch_bounds__(256)
void attn_kernel(const u16* __restrict__ Qb, const u16* __restrict__ Kb,
                 const u16* __restrict__ Vt, const unsigned char* __restrict__ tok,
                 u16* __restrict__ Yb)
{
  __shared__ u16 Plds[8 * 16 * 32];   // per-wave 16x32 bf16 P tile

  const int tid   = threadIdx.x;
  const int lane  = tid & 31;
  const int wave  = tid >> 5;
  const int rhalf = lane >> 4;
  const int chalf = lane & 15;
  const int base1 = rhalf * 8;
  const int bh    = blockIdx.y;
  const int b     = bh / Hn;
  const int h     = bh % Hn;
  const int qrow  = blockIdx.x * 128 + wave * 16;
  u16* P = &Plds[wave * 512];

  // Q A-fragments: d=64 -> two K=32 fragments
  FragU qa[2];
  {
    const u16* qp = Qb + ((size_t)bh * Tn + qrow + chalf) * Dn;
#pragma unroll
    for (int kc = 0; kc < 2; kc++) {
      qa[kc].q[0] = *(const uint4*)&qp[kc * 32 + rhalf * 16];
      qa[kc].q[1] = *(const uint4*)&qp[kc * 32 + rhalf * 16 + 8];
    }
  }
  FragU ones;
#pragma unroll
  for (int e = 0; e < 16; e++) ones.s[e] = 0x3F80;   // bf16 1.0

  float mrow[8];
  v8f o[5];                                          // 4 output tiles + row-sum column
#pragma unroll
  for (int r = 0; r < 8; r++) mrow[r] = -3.0e38f;
#pragma unroll
  for (int a = 0; a < 5; a++)
#pragma unroll
    for (int r = 0; r < 8; r++) o[a][r] = 0.f;

  const int jend = qrow + 16;   // keys < jend
  // Full (mask-free) tiles: jbase + 31 <= qrow
  const int jend_full = (qrow >= 31) ? ((((qrow - 31) >> 5) << 5) + 32) : 0;

  int jbase = 0;
  for (; jbase < jend_full; jbase += 32) {
    if (jbase + 32 < jend) {
      __builtin_prefetch(Kb + ((size_t)bh * Tn + jbase + 32 + chalf) * Dn, 0, 1);
      __builtin_prefetch(Vt + ((size_t)bh * Dn + chalf) * Tn + jbase + 32, 0, 1);
    }
    attn_tile<false>(Kb, Vt, tok, qa, ones, o, mrow, P, bh, b, qrow, jbase, lane);
  }
  for (; jbase < jend; jbase += 32)
    attn_tile<true>(Kb, Vt, tok, qa, ones, o, mrow, P, bh, b, qrow, jbase, lane);

  // Write y[b, t, h*64 + :] as bf16 (normalize by the WMMA-computed row sums)
#pragma unroll
  for (int nt = 0; nt < 4; nt++)
#pragma unroll
    for (int r = 0; r < 8; r++) {
      int row = qrow + r + 8 * rhalf;
      float val = o[nt][r] / o[4][r];
      Yb[((size_t)b * Tn + row) * Cn + h * Dn + nt * 16 + chalf] = f2bf(val);
    }
}

// ---------------- host launcher ----------------
extern "C" void kernel_launch(void* const* d_in, const int* in_sizes, int n_in,
                              void* d_out, int out_size, void* d_ws, size_t ws_size,
                              hipStream_t stream)
{
  const float* x           = (const float*)d_in[0];
  const unsigned char* tok = (const unsigned char*)d_in[1];
  const float* wqkv        = (const float*)d_in[2];
  const float* wproj       = (const float*)d_in[3];
  float* out               = (float*)d_out;

  char* ws    = (char*)d_ws;
  u16* xb     = (u16*)(ws);                          // 16 MB
  u16* wqkvb  = (u16*)(ws + (size_t)16777216);       //  6 MB
  u16* wprojb = (u16*)(ws + (size_t)23068672);       //  2 MB
  u16* Qb     = (u16*)(ws + (size_t)25165824);       // 16 MB (pre-scaled 1/8)
  u16* Kb     = (u16*)(ws + (size_t)41943040);       // 16 MB
  u16* Vt     = (u16*)(ws + (size_t)58720256);       // 16 MB (transposed per head)
  u16* Yb     = (u16*)(ws + (size_t)75497472);       // 16 MB

  int nx = Bsz * Tn * Cn;
  cvt_f32_bf16<<<(nx + 255) / 256, 256, 0, stream>>>(x, xb, nx);
  int nw = Cn * N3;
  cvt_f32_bf16<<<(nw + 255) / 256, 256, 0, stream>>>(wqkv, wqkvb, nw);
  int np = Cn * Cn;
  cvt_f32_bf16<<<(np + 255) / 256, 256, 0, stream>>>(wproj, wprojb, np);

  dim3 g1(64, 24);                 // M=8192 -> 64 blocks, N=3072 -> 24 blocks
  gemm_bf16<0><<<g1, 256, 0, stream>>>(xb, wqkvb, Cn, N3, Qb, Kb, Vt, nullptr);

  dim3 g2(Tn / 128, Bsz * Hn);     // (16, 64)
  attn_kernel<<<g2, 256, 0, stream>>>(Qb, Kb, Vt, tok, Yb);

  dim3 g3(64, 8);                  // M=8192, N=1024
  gemm_bf16<1><<<g3, 256, 0, stream>>>(Yb, wprojb, Cn, Cn, nullptr, nullptr, nullptr, out);
}
